// RNNmodel_71536975282182
// MI455X (gfx1250) — matrix operand order repaired and checked by reading, not compiled, not run on hardware
//
#include <hip/hip_runtime.h>

typedef __attribute__((ext_vector_type(16))) _Float16 v16h;
typedef __attribute__((ext_vector_type(8)))  _Float16 v8h;
typedef __attribute__((ext_vector_type(8)))  float    v8f;

#define HH 1024      // hidden
#define GATES 4096   // 4*H
#define BB 32        // batch
#define TT 1024      // seq len
#define II 512       // input size
#define OO 256       // out dim

__device__ __forceinline__ v16h cat8(v8h lo, v8h hi) {
    v16h r;
#pragma unroll
    for (int i = 0; i < 8; ++i) { r[i] = lo[i]; r[i + 8] = hi[i]; }
    return r;
}

__device__ __forceinline__ float sigf(float x) { return 1.0f / (1.0f + __expf(-x)); }

// --- CDNA5 async global->LDS copies (ASYNCcnt-tracked, no VGPR staging) ----
__device__ __forceinline__ uint32_t lds_off(const void* p) {
    // flat LDS address = {aperture[63:32], lds_byte_offset[31:0]}
    return (uint32_t)(uintptr_t)p;
}
__device__ __forceinline__ void async_b128(uint32_t lds, const void* g) {
    asm volatile("global_load_async_to_lds_b128 %0, %1, off"
                 :: "v"(lds), "v"((uint64_t)(uintptr_t)g) : "memory");
}
__device__ __forceinline__ void async_b64(uint32_t lds, const void* g) {
    asm volatile("global_load_async_to_lds_b64 %0, %1, off"
                 :: "v"(lds), "v"((uint64_t)(uintptr_t)g) : "memory");
}

// ---------------------------------------------------------------------------
// Generic f16 GEMM: C[M,N] = act(A[M,K] @ W[N,K]^T + bias)
// OMODE 0: store f16; OMODE 1: tanh, store f16; OMODE 2: store f32 with
// [T,B]-row -> [B,T]-row permutation (final output).
// Tile 128x128x32, 256 threads = 8 waves, wave w owns rows [16w,16w+16).
// ---------------------------------------------------------------------------
template <int OMODE>
__global__ __launch_bounds__(256) void gemm_wmma_f16(
    const _Float16* __restrict__ A, const _Float16* __restrict__ W,
    const float* __restrict__ bias, void* __restrict__ out,
    int M, int N, int K, int Tdim) {
    (void)Tdim;
    const int tid = threadIdx.x;
    const int w = tid >> 5, lane = tid & 31;
    const int m0 = blockIdx.y * 128, n0 = blockIdx.x * 128;

    __shared__ _Float16 lA[128][40];
    __shared__ _Float16 lB[128][40];

    v8f acc[8] = {};

    const int lrow = tid >> 1;            // 0..127
    const int seg = (tid & 1) * 16;       // 0 or 16
    const _Float16* gA = A + (size_t)(m0 + lrow) * K + seg;
    const _Float16* gB = W + (size_t)(n0 + lrow) * K + seg;

    const int r = w * 16 + (lane & 15);
    const int kb = (lane >> 4) * 8;
    const int kb2 = (lane >> 4) * 16;

    for (int k0 = 0; k0 < K; k0 += 32) {
        v8h a0 = *(const v8h*)(gA + k0);
        v8h a1 = *(const v8h*)(gA + k0 + 8);
        v8h b0 = *(const v8h*)(gB + k0);
        v8h b1 = *(const v8h*)(gB + k0 + 8);
        if (k0 + 32 < K) {  // pull next chunk toward the WGP (global_prefetch_b8)
            __builtin_prefetch(gA + k0 + 32, 0, 3);
            __builtin_prefetch(gB + k0 + 32, 0, 3);
        }
        __syncthreads();
        *(v8h*)&lA[lrow][seg] = a0; *(v8h*)&lA[lrow][seg + 8] = a1;
        *(v8h*)&lB[lrow][seg] = b0; *(v8h*)&lB[lrow][seg + 8] = b1;
        __syncthreads();

        v16h af = cat8(*(const v8h*)&lA[r][kb], *(const v8h*)&lA[r][kb + 16]);
#pragma unroll
        for (int nt = 0; nt < 8; ++nt) {
            const int c = nt * 16 + (lane & 15);
            v16h bf = cat8(*(const v8h*)&lB[c][kb], *(const v8h*)&lB[c][kb + 16]);
            acc[nt] = __builtin_amdgcn_wmma_f32_16x16x32_f16(
                false, af, false, bf, (short)0, acc[nt], false, false);
        }
    }

#pragma unroll
    for (int nt = 0; nt < 8; ++nt) {
        const int n = n0 + nt * 16 + (lane & 15);
        const float bv = bias[n];
#pragma unroll
        for (int v = 0; v < 8; ++v) {
            const int m = m0 + w * 16 + v + (lane >> 4) * 8;
            float val = acc[nt][v] + bv;
            if (OMODE == 1) val = tanhf(val);
            if (OMODE == 2) {
                const int b = m & (BB - 1);
                const int t = m >> 5;
                ((float*)out)[((size_t)b * Tdim + t) * N + n] = val;
            } else {
                ((_Float16*)out)[(size_t)m * N + n] = (_Float16)val;
            }
        }
    }
}

// ---------------------------------------------------------------------------
// Fused LSTM recurrent step: gates = xp[t] + h_prev @ Whh^T, then cell.
// Grid: 32 blocks, block owns hidden cols j0..j0+31 for ALL 4 gates
// (128 W rows). 8 waves: wave = (mtile 0..1) x (gate 0..3). K loop = 1024.
// Tiles are staged with double-buffered GLOBAL_LOAD_ASYNC_TO_LDS copies:
// every thread issues exactly 3 async ops per chunk (2xB128 W, 1xB64 h), so
// `s_wait_asynccnt 3` after issuing chunk k+1 guarantees chunk k has landed.
// ---------------------------------------------------------------------------
__global__ __launch_bounds__(256) void lstm_step_wmma(
    const _Float16* __restrict__ xpt,    // [32, 4096] this timestep
    const _Float16* __restrict__ hprev,  // [32, 1024]
    const _Float16* __restrict__ Whh,    // [4096, 1024] f16
    float* __restrict__ cst,             // [32, 1024]
    _Float16* __restrict__ hout) {       // [32, 1024]
    const int tid = threadIdx.x;
    const int w = tid >> 5, lane = tid & 31;
    const int j0 = blockIdx.x * 32;
    const int g = w & 3, mt = w >> 2;

    __shared__ _Float16 lH[2][32][40];
    __shared__ _Float16 lW[2][128][40];
    __shared__ float lG[32][132];

    v8f acc[2] = {};

    // W-tile copy job: thread -> (row 0..127, 16-half segment)
    const int lrow = tid >> 1;
    const int seg = (tid & 1) * 16;
    const int wgate = lrow >> 5, wj = lrow & 31;
    const _Float16* gW = Whh + (size_t)(wgate * HH + j0 + wj) * HH + seg;
    // h-tile copy job: thread -> (row 0..31, 4-half segment)
    const int hrow = tid >> 3;
    const int hcol = (tid & 7) * 4;
    const _Float16* gH = hprev + (size_t)hrow * HH + hcol;

    const int r = mt * 16 + (lane & 15);
    const int kb = (lane >> 4) * 8;
    const int kb2 = (lane >> 4) * 16;

    const int NK = HH / 32;

    // prologue: stage chunk 0 into buffer 0
    async_b128(lds_off(&lW[0][lrow][seg]), gW);
    async_b128(lds_off(&lW[0][lrow][seg + 8]), gW + 8);
    async_b64(lds_off(&lH[0][hrow][hcol]), gH);

    for (int kc = 0; kc < NK; ++kc) {
        const int buf = kc & 1;
        if (kc + 1 < NK) {
            const int k1 = (kc + 1) * 32;
            const int nb = buf ^ 1;
            async_b128(lds_off(&lW[nb][lrow][seg]), gW + k1);
            async_b128(lds_off(&lW[nb][lrow][seg + 8]), gW + k1 + 8);
            async_b64(lds_off(&lH[nb][hrow][hcol]), gH + k1);
            asm volatile("s_wait_asynccnt 0x3" ::: "memory");
        } else {
            asm volatile("s_wait_asynccnt 0x0" ::: "memory");
        }
        __syncthreads();  // all waves' chunk-kc tiles are resident

        v16h af = cat8(*(const v8h*)&lH[buf][r][kb],
                       *(const v8h*)&lH[buf][r][kb + 16]);
#pragma unroll
        for (int nt = 0; nt < 2; ++nt) {
            const int cc = g * 32 + nt * 16 + (lane & 15);
            v16h bf = cat8(*(const v8h*)&lW[buf][cc][kb],
                           *(const v8h*)&lW[buf][cc][kb + 16]);
            acc[nt] = __builtin_amdgcn_wmma_f32_16x16x32_f16(
                false, af, false, bf, (short)0, acc[nt], false, false);
        }
        __syncthreads();  // done reading buf before it is overwritten
    }

#pragma unroll
    for (int nt = 0; nt < 2; ++nt)
#pragma unroll
        for (int v = 0; v < 8; ++v) {
            const int m = mt * 16 + v + (lane >> 4) * 8;
            const int n = g * 32 + nt * 16 + (lane & 15);
            lG[m][n] = acc[nt][v];
        }
    __syncthreads();

#pragma unroll
    for (int it = 0; it < 4; ++it) {
        const int p = tid + it * 256;   // 1024 (m,j) pairs
        const int m = p >> 5, jj = p & 31, j = j0 + jj;
        const size_t xb = (size_t)m * GATES + j;
        float iv = lG[m][jj]      + (float)xpt[xb];
        float fv = lG[m][32 + jj] + (float)xpt[xb + HH];
        float gv = lG[m][64 + jj] + (float)xpt[xb + 2 * HH];
        float ov = lG[m][96 + jj] + (float)xpt[xb + 3 * HH];
        float i_ = sigf(iv), f_ = sigf(fv), g_ = tanhf(gv), o_ = sigf(ov);
        const size_t ci = (size_t)m * HH + j;
        float cn = f_ * cst[ci] + i_ * g_;
        cst[ci] = cn;
        hout[ci] = (_Float16)(o_ * tanhf(cn));
    }
}

// ---------------------------------------------------------------------------
// Helpers: f32 -> f16 convert, and x [B,T,I] -> f16 [T,B,I] transpose-convert
// ---------------------------------------------------------------------------
__global__ void f32_to_f16_k(const float* __restrict__ in,
                             _Float16* __restrict__ out, size_t n) {
    size_t i = (size_t)blockIdx.x * blockDim.x + threadIdx.x;
    const size_t stride = (size_t)gridDim.x * blockDim.x;
    for (; i < n; i += stride) out[i] = (_Float16)in[i];
}

__global__ void x_transpose_f16_k(const float* __restrict__ in,
                                  _Float16* __restrict__ out) {
    const size_t n = (size_t)BB * TT * II;
    size_t i = (size_t)blockIdx.x * blockDim.x + threadIdx.x;
    const size_t stride = (size_t)gridDim.x * blockDim.x;
    for (; i < n; i += stride) {
        const int ii = (int)(i & (II - 1));
        const int b = (int)((i >> 9) & (BB - 1));
        const int t = (int)(i >> 14);
        out[i] = (_Float16)in[((size_t)b * TT + t) * II + ii];
    }
}

// ---------------------------------------------------------------------------
extern "C" void kernel_launch(void* const* d_in, const int* in_sizes, int n_in,
                              void* d_out, int out_size, void* d_ws, size_t ws_size,
                              hipStream_t stream) {
    (void)in_sizes; (void)n_in; (void)out_size; (void)ws_size;
    const float* x    = (const float*)d_in[0];
    const float* Wih0 = (const float*)d_in[1];
    const float* Whh0 = (const float*)d_in[2];
    const float* b0   = (const float*)d_in[3];
    const float* Wih1 = (const float*)d_in[4];
    const float* Whh1 = (const float*)d_in[5];
    const float* b1   = (const float*)d_in[6];
    const float* Wr1  = (const float*)d_in[7];
    const float* br1  = (const float*)d_in[8];
    const float* Wr2  = (const float*)d_in[9];
    const float* br2  = (const float*)d_in[10];

    char* ws = (char*)d_ws;
    size_t off = 0;
    auto alloc = [&](size_t bytes) -> void* {
        void* p = ws + off;
        off = (off + bytes + 255) & ~(size_t)255;
        return p;
    };
    const size_t MROWS = (size_t)TT * BB;  // 32768 token rows
    _Float16* xp    = (_Float16*)alloc(MROWS * GATES * 2);          // 256 MB
    _Float16* xT    = (_Float16*)alloc(MROWS * II * 2);             // 32 MB
    _Float16* hbuf  = (_Float16*)alloc((size_t)(TT + 1) * BB * HH * 2); // 67 MB
    _Float16* tmp   = (_Float16*)alloc(MROWS * HH * 2);             // 64 MB
    _Float16* wih0f = (_Float16*)alloc((size_t)GATES * II * 2);
    _Float16* whh0f = (_Float16*)alloc((size_t)GATES * HH * 2);
    _Float16* wih1f = (_Float16*)alloc((size_t)GATES * HH * 2);
    _Float16* whh1f = (_Float16*)alloc((size_t)GATES * HH * 2);
    _Float16* wr1f  = (_Float16*)alloc((size_t)HH * HH * 2);
    _Float16* wr2f  = (_Float16*)alloc((size_t)OO * HH * 2);
    float*    cst   = (float*)alloc((size_t)BB * HH * 4);

    // ---- weight / input conversion to f16 ----
    f32_to_f16_k<<<1024, 256, 0, stream>>>(Wih0, wih0f, (size_t)GATES * II);
    f32_to_f16_k<<<1024, 256, 0, stream>>>(Whh0, whh0f, (size_t)GATES * HH);
    f32_to_f16_k<<<1024, 256, 0, stream>>>(Wih1, wih1f, (size_t)GATES * HH);
    f32_to_f16_k<<<1024, 256, 0, stream>>>(Whh1, whh1f, (size_t)GATES * HH);
    f32_to_f16_k<<<1024, 256, 0, stream>>>(Wr1, wr1f, (size_t)HH * HH);
    f32_to_f16_k<<<1024, 256, 0, stream>>>(Wr2, wr2f, (size_t)OO * HH);
    x_transpose_f16_k<<<2048, 256, 0, stream>>>(x, xT);

    const dim3 blk(256);
    // ---- layer 0: xp = xT @ Wih0^T + b0 ----
    gemm_wmma_f16<0><<<dim3(GATES / 128, MROWS / 128), blk, 0, stream>>>(
        xT, wih0f, b0, xp, (int)MROWS, GATES, II, 0);
    hipMemsetAsync(cst, 0, (size_t)BB * HH * 4, stream);
    hipMemsetAsync(hbuf, 0, (size_t)BB * HH * 2, stream);
    for (int t = 0; t < TT; ++t) {
        lstm_step_wmma<<<32, blk, 0, stream>>>(
            xp + (size_t)t * BB * GATES, hbuf + (size_t)t * BB * HH, whh0f,
            cst, hbuf + (size_t)(t + 1) * BB * HH);
    }
    // ---- layer 1: xp = h1 @ Wih1^T + b1 (h1 = hbuf slices 1..T) ----
    gemm_wmma_f16<0><<<dim3(GATES / 128, MROWS / 128), blk, 0, stream>>>(
        hbuf + (size_t)BB * HH, wih1f, b1, xp, (int)MROWS, GATES, HH, 0);
    hipMemsetAsync(cst, 0, (size_t)BB * HH * 4, stream);
    hipMemsetAsync(hbuf, 0, (size_t)BB * HH * 2, stream);
    for (int t = 0; t < TT; ++t) {
        lstm_step_wmma<<<32, blk, 0, stream>>>(
            xp + (size_t)t * BB * GATES, hbuf + (size_t)t * BB * HH, whh1f,
            cst, hbuf + (size_t)(t + 1) * BB * HH);
    }
    // ---- head: tmp = tanh(h2 @ Wr1^T + br1); out = tmp @ Wr2^T + br2 ----
    gemm_wmma_f16<1><<<dim3(HH / 128, MROWS / 128), blk, 0, stream>>>(
        hbuf + (size_t)BB * HH, wr1f, br1, tmp, (int)MROWS, HH, HH, 0);
    gemm_wmma_f16<2><<<dim3(OO / 128, MROWS / 128), blk, 0, stream>>>(
        tmp, wr2f, br2, d_out, (int)MROWS, OO, HH, TT);
}
